// ForwardEulerSimulator_23046794510947
// MI455X (gfx1250) — matrix operand order repaired and checked by reading, not compiled, not run on hardware
//
#include <hip/hip_runtime.h>

// ---- problem constants (from reference) ----
#define N_X    32
#define N_U    16
#define N_FEAT 512
#define BATCH  512
#define SEQLEN 2048
#define TS_C   0.01f

typedef __attribute__((ext_vector_type(16))) __bf16 v16bf;
typedef __attribute__((ext_vector_type(8)))  float  v8f;

union AB { v16bf v; unsigned int u[8]; };

// float -> bf16 bits via native hardware convert (v_cvt_pk_bf16_f32)
__device__ __forceinline__ unsigned short f2bf16(float f) {
    __bf16 b = (__bf16)f;
    return __builtin_bit_cast(unsigned short, b);
}

// Hardware V_TANH_F32 on gfx1250; branch-free fallback otherwise.
__device__ __forceinline__ float fast_tanh(float x) {
#if __has_builtin(__builtin_amdgcn_tanhf)
    return __builtin_amdgcn_tanhf(x);
#elif __has_builtin(__builtin_amdgcn_tanh_f32)
    return __builtin_amdgcn_tanh_f32(x);
#else
    const float LOG2E_X2 = 2.885390081777927f;   // 2*log2(e)
    float t = __builtin_fabsf(x);
#if __has_builtin(__builtin_amdgcn_exp2f) && __has_builtin(__builtin_amdgcn_rcpf)
    float e = __builtin_amdgcn_exp2f(t * LOG2E_X2);        // exp(2|x|), inf ok
    float r = 1.0f - 2.0f * __builtin_amdgcn_rcpf(e + 1.0f);
#else
    float e = __expf(t * 2.0f);
    float r = 1.0f - 2.0f / (e + 1.0f);
#endif
    return __builtin_copysignf(r, x);
#endif
}

// LDS strides (in 32-bit pair units) — padded to avoid bank-conflict pathologies
#define W1_STRIDE 36    // 72 bf16 per feature column n (k = 0..63 used, zero-padded past 47)
#define W2_STRIDE 260   // 520 bf16 per output column n (k = 0..511)
#define H_STRIDE  260   // 520 bf16 per batch row m   (k = 0..511)
#define XU_STRIDE 36    // 72 bf16 per batch row m    (k = 0..63 used: 32 x | 16 u | 16 zero-pad)

__global__ __launch_bounds__(256)
void fwd_euler_rnn_kernel(const float* __restrict__ x0,   // (512, 1, 32)
                          const float* __restrict__ u,    // (512, 2048, 16)
                          const float* __restrict__ W1,   // (48, 512)
                          const float* __restrict__ b1,   // (512)
                          const float* __restrict__ W2,   // (512, 32)
                          const float* __restrict__ b2,   // (32)
                          float* __restrict__ out)        // (512, 2048, 32)
{
    // LDS working set ~144 KB (<= 320 KB/WGP)
    __shared__ unsigned int W1s[N_FEAT * W1_STRIDE];   // W1 transposed: [n][k] bf16 pairs
    __shared__ unsigned int W2s[N_X * W2_STRIDE];      // W2 transposed: [n][k] bf16 pairs
    __shared__ unsigned int h_s[16 * H_STRIDE];        // activations:  [m][k] bf16 pairs
    __shared__ unsigned int xu [16 * XU_STRIDE];       // A operand:    [m][x|u|0] bf16 pairs
    __shared__ float dxp[8 * 16 * N_X];                // per-wave GEMM2 partials
    __shared__ float b1l[N_FEAT];
    __shared__ float b2l[N_X];

    const int tid    = threadIdx.x;
    const int wave   = tid >> 5;
    const int lane   = tid & 31;
    const int lanehi = (lane >> 4) & 1;     // 0: lanes 0-15, 1: lanes 16-31
    const int lrow   = lane & 15;
    const int m0     = blockIdx.x * 16;     // batch-tile base row

    unsigned short* h16  = reinterpret_cast<unsigned short*>(h_s);
    unsigned short* xu16 = reinterpret_cast<unsigned short*>(xu);

    // ---------------- init: stage weights/bias/state into LDS ----------------
    // W1s[n][pair p] , k = 2p (zero-pad k >= 48)
    for (int idx = tid; idx < N_FEAT * 32; idx += 256) {
        int n = idx >> 5, p = idx & 31, k = p << 1;
        unsigned int lo = (k     < 48) ? f2bf16(W1[(size_t)k       * N_FEAT + n]) : 0u;
        unsigned int hi = (k + 1 < 48) ? f2bf16(W1[(size_t)(k + 1) * N_FEAT + n]) : 0u;
        W1s[n * W1_STRIDE + p] = lo | (hi << 16);
    }
    // W2s[n][pair p] , k = 2p
    for (int idx = tid; idx < N_X * 256; idx += 256) {
        int n = idx >> 8, p = idx & 255, k = p << 1;
        unsigned int lo = f2bf16(W2[(size_t)k       * N_X + n]);
        unsigned int hi = f2bf16(W2[(size_t)(k + 1) * N_X + n]);
        W2s[n * W2_STRIDE + p] = lo | (hi << 16);
    }
    for (int idx = tid; idx < N_FEAT; idx += 256) b1l[idx] = b1[idx];
    if (tid < N_X) b2l[tid] = b2[tid];
    // [x,u,0] A operand for t = 0
    for (int idx = tid; idx < 16 * XU_STRIDE; idx += 256) {
        int m = idx / XU_STRIDE, p = idx % XU_STRIDE;
        unsigned int val = 0u;
        if (p < 16) {              // x part: k = 2p, 2p+1  (k in 0..31)
            int n = p << 1;
            val = (unsigned int)f2bf16(x0[(size_t)(m0 + m) * N_X + n]) |
                  ((unsigned int)f2bf16(x0[(size_t)(m0 + m) * N_X + n + 1]) << 16);
        } else if (p < 24) {       // u part: k = 32 + 2(p-16)
            int j = (p - 16) << 1;
            const float* up = u + ((size_t)(m0 + m) * SEQLEN) * N_U;
            val = (unsigned int)f2bf16(up[j]) | ((unsigned int)f2bf16(up[j + 1]) << 16);
        }
        xu[idx] = val;             // p >= 24 -> zero K-pad (k 48..63)
    }
    __syncthreads();

    // f32 state held in registers: this thread owns elements tid and tid+256 of
    // the 16x32 state tile (idx -> m = idx>>5, n = idx&31).
    const int mA = tid >> 5,          nA = tid & 31;
    const int mB = (tid + 256) >> 5,  nB = tid & 31;    // (tid+256)&31 == tid&31
    float xA = x0[(size_t)(m0 + mA) * N_X + nA];
    float xB = x0[(size_t)(m0 + mB) * N_X + nB];
    const float b2A = b2l[nA];        // same n for both owned elements

    // Per-tile bias splat as WMMA C-operand init (D = A*B + C): b1[n] is
    // constant over the whole time loop for this lane's column.
    v8f cinit[4];
    #pragma unroll
    for (int i = 0; i < 4; ++i) {
        float bias = b1l[(wave * 4 + i) * 16 + lrow];
        #pragma unroll
        for (int r = 0; r < 8; ++r) cinit[i][r] = bias;
    }

    // ---------------- time loop (sequential recurrence) ----------------
    for (int t = 0; t < SEQLEN; ++t) {
        // ===== GEMM1: h = tanh([x,u] @ W1 + b1), M=16, K=64(padded), N=512 =====
        // A operand (16-bit A layout, ISA 7.12.2): per 32-K chunk, lanes 0-15 read
        // k pairs {0..3, 8..11}, lanes 16-31 read {4..7, 12..15} (pair units).
        AB a0, a1;
        #pragma unroll
        for (int p = 0; p < 8; ++p) {
            int off = p + ((p >= 4) ? 4 : 0) + (lanehi ? 4 : 0);
            a0.u[p] = xu[lrow * XU_STRIDE + off];        // K chunk 0 (k 0..31)
            a1.u[p] = xu[lrow * XU_STRIDE + 16 + off];   // K chunk 1 (k 32..63)
        }
        #pragma unroll
        for (int i = 0; i < 4; ++i) {
            int nt = wave * 4 + i;
            int n  = nt * 16 + lrow;
            AB bA, bB;   // B operand: same k pairs as A (lanes 0-15 {0..3, 8..11}, lanes 16-31 {4..7, 12..15}) per chunk
            #pragma unroll
            for (int p = 0; p < 8; ++p) {
                int ob = p + ((p >= 4) ? 4 : 0) + (lanehi ? 4 : 0);
                bA.u[p] = W1s[n * W1_STRIDE + ob];
                bB.u[p] = W1s[n * W1_STRIDE + 16 + ob];
            }
            v8f acc = __builtin_amdgcn_wmma_f32_16x16x32_bf16(false, a0.v, false, bA.v,
                                                              (short)0, cinit[i], false, false);
            acc = __builtin_amdgcn_wmma_f32_16x16x32_bf16(false, a1.v, false, bB.v,
                                                          (short)0, acc, false, false);
            #pragma unroll
            for (int r = 0; r < 8; ++r) {
                int M = r + (lanehi << 3);
                h16[M * (H_STRIDE * 2) + n] = f2bf16(fast_tanh(acc[r]));
            }
        }
        __syncthreads();

        // ===== GEMM2: dx = h @ W2, M=16, K=512, N=32; K split across 8 waves =====
        #pragma unroll
        for (int nt2 = 0; nt2 < 2; ++nt2) {
            v8f acc = {};
            #pragma unroll
            for (int kk = 0; kk < 2; ++kk) {
                int kc = wave * 2 + kk;                 // 32-wide K chunk index
                AB a, b;
                #pragma unroll
                for (int p = 0; p < 8; ++p) {
                    int oa = kc * 16 + p + ((p >= 4) ? 4 : 0) + (lanehi ? 4 : 0);
                    a.u[p] = h_s[lrow * H_STRIDE + oa];
                int ob = kc * 16 + p + ((p >= 4) ? 4 : 0) + (lanehi ? 4 : 0);
                    b.u[p] = W2s[(nt2 * 16 + lrow) * W2_STRIDE + ob];
                }
                acc = __builtin_amdgcn_wmma_f32_16x16x32_bf16(false, a.v, false, b.v,
                                                              (short)0, acc, false, false);
            }
            #pragma unroll
            for (int r = 0; r < 8; ++r) {
                int M = r + (lanehi << 3);
                dxp[wave * (16 * N_X) + M * N_X + nt2 * 16 + lrow] = acc[r];
            }
        }
        __syncthreads();

        // ===== reduce partials, Euler update, emit state, stage next [x,u] =====
        {
            float sA = 0.f, sB = 0.f;
            #pragma unroll
            for (int w = 0; w < 8; ++w) {
                sA += dxp[w * (16 * N_X) + tid];
                sB += dxp[w * (16 * N_X) + tid + 256];
            }
            xA += TS_C * (sA + b2A);
            xB += TS_C * (sB + b2A);
            out[((size_t)(m0 + mA) * SEQLEN + t) * N_X + nA] = xA;
            out[((size_t)(m0 + mB) * SEQLEN + t) * N_X + nB] = xB;
            xu16[mA * (XU_STRIDE * 2) + nA] = f2bf16(xA);
            xu16[mB * (XU_STRIDE * 2) + nB] = f2bf16(xB);
        }
        if (t + 1 < SEQLEN) {
            int m = tid >> 4, j = tid & 15;   // 256 threads -> 16x16 u elements
            float uv = u[((size_t)(m0 + m) * SEQLEN + (t + 1)) * N_U + j];
            xu16[m * (XU_STRIDE * 2) + 32 + j] = f2bf16(uv);
        }
        if (t + 2 < SEQLEN && tid < 16) {     // hint next-next u row into cache
            __builtin_prefetch(&u[((size_t)(m0 + tid) * SEQLEN + (t + 2)) * N_U], 0, 0);
        }
        __syncthreads();
    }
}

extern "C" void kernel_launch(void* const* d_in, const int* in_sizes, int n_in,
                              void* d_out, int out_size, void* d_ws, size_t ws_size,
                              hipStream_t stream) {
    (void)in_sizes; (void)n_in; (void)out_size; (void)d_ws; (void)ws_size;
    const float* x0 = (const float*)d_in[0];  // x0_batch (512,1,32)
    const float* u  = (const float*)d_in[1];  // u_batch  (512,2048,16)
    const float* W1 = (const float*)d_in[2];  // (48,512)
    const float* b1 = (const float*)d_in[3];  // (512)
    const float* W2 = (const float*)d_in[4];  // (512,32)
    const float* b2 = (const float*)d_in[5];  // (32)
    float* out = (float*)d_out;               // (512,2048,32)

    dim3 grid(BATCH / 16);    // 32 workgroups, one 16-row batch tile each
    dim3 block(256);          // 8 wave32 waves
    fwd_euler_rnn_kernel<<<grid, block, 0, stream>>>(x0, u, W1, b1, W2, b2, out);
}
